// SwinTransformerBlock_81372450390651
// MI455X (gfx1250) — matrix operand order repaired and checked by reading, not compiled, not run on hardware
//
#include <hip/hip_runtime.h>
#include <hip/hip_bf16.h>
#include <math.h>

typedef __attribute__((ext_vector_type(16))) _Float16 v16h;
typedef __attribute__((ext_vector_type(8)))  _Float16 v8h;
typedef __attribute__((ext_vector_type(8)))  float    v8f;

#define BB    16
#define HH    112
#define WWI   112
#define CC    96
#define NHEAD 3
#define HDIM  32
#define WSZ   7
#define NTOK  49
#define SHFT  3
#define HIDN  384
#define FEPS  1e-6f

// ---------------- WMMA operand helpers (gfx1250 wave32, 16x16x32 f16) ---------
// A 16x32 f16: lane L: m = L%16, g = L/16; halves i: K = i + 16g
// -> two contiguous 8-half chunks at k0+16g and k0+16g+8.
__device__ __forceinline__ v16h ld_a_rm(const _Float16* base, int ld, int m0, int k0, int lane) {
    int m = m0 + (lane & 15);
    int g = lane >> 4;
    const _Float16* p = base + m * ld + k0 + 16 * g;
    v8h lo = *(const v8h*)p;
    v8h hi = *(const v8h*)(p + 8);
    v16h r;
#pragma unroll
    for (int i = 0; i < 8; i++) { r[i] = lo[i]; r[i + 8] = hi[i]; }
    return r;
}

// B 32x16 f16 from weight stored row-major W[N][K] (computes A @ W^T):
// lane L: n = L%16, g = L/16; halves i: K = k0 + i + 16g -> one contiguous 16-half chunk.
__device__ __forceinline__ v16h ld_b_wt(const _Float16* W, int ld, int n0, int k0, int lane) {
    int n = n0 + (lane & 15);
    int g = lane >> 4;
    return *(const v16h*)(W + n * ld + k0 + 16 * g);
}

// A operand gathered transposed from row-major buf: A[m][k] = base[k*ld + col0 + m]
__device__ __forceinline__ v16h ld_a_tr(const _Float16* base, int ld, int col0, int k0, int lane) {
    int m = lane & 15;
    int g = lane >> 4;
    v16h r;
#pragma unroll
    for (int i = 0; i < 16; i++) {
        int k = k0 + i + 16 * g;
        r[i] = base[k * ld + col0 + m];
    }
    return r;
}

// B operand gathered from row-major buf B[k][n] = base[k*ld + n0 + n]
__device__ __forceinline__ v16h ld_b_rm(const _Float16* base, int ld, int n0, int k0, int lane) {
    int n = n0 + (lane & 15);
    int g = lane >> 4;
    v16h r;
#pragma unroll
    for (int i = 0; i < 16; i++) r[i] = base[(k0 + i + 16 * g) * ld + n];
    return r;
}

__device__ __forceinline__ v8f wmma16(v16h a, v16h b, v8f c) {
    return __builtin_amdgcn_wmma_f32_16x16x32_f16(false, a, false, b, (short)0, c, false, false);
}

// ---------------- LDS layout for attention kernel (62592 B) ------------------
#define O_XW   0        // xwin f16 64x96 (12288) -> later q_h f16
#define O_QF   12288    // staging f32 / qf f32 64x96 (24576)
#define O_KH   12288    // k_h f16 64x96 (after qf dead)
#define O_VH   24576    // v_h f16 64x96
#define O_KF   36864    // kf f32 64x96 (24576)
#define O_KV   36864    // kv_h f16 3x32x32 (6144)  (after kf dead)
#define O_DWW  43008    // dwc_w f32 800 (3200)
#define O_DWB  46208    // dwc_b f32 32 (128)
#define O_OUT  46336    // out_h f16 64x96 (12288)
#define O_KM   61440    // rsc f32 96 (P0..P1) -> kmean f32 96 (P4..)
#define O_Z    61824    // z f32 192 (768)
#define SMEM_ATT 62592

__global__ __launch_bounds__(256)
void attn_kernel(const float* __restrict__ x,
                 const float* __restrict__ n1g, const float* __restrict__ n1b,
                 const _Float16* __restrict__ Wqkv, const float* __restrict__ bqkv,
                 const float* __restrict__ scale_p,
                 const float* __restrict__ dwc_w, const float* __restrict__ dwc_b,
                 const _Float16* __restrict__ Wproj, const float* __restrict__ bproj,
                 float* __restrict__ x2) {
    __shared__ char smem[SMEM_ATT];
    _Float16* xwin  = (_Float16*)(smem + O_XW);
    _Float16* qh    = (_Float16*)(smem + O_XW);
    float*    qf    = (float*)(smem + O_QF);
    _Float16* kh    = (_Float16*)(smem + O_KH);
    _Float16* vh    = (_Float16*)(smem + O_VH);
    float*    kf    = (float*)(smem + O_KF);
    _Float16* kvh   = (_Float16*)(smem + O_KV);
    float*    wdw   = (float*)(smem + O_DWW);
    float*    bdw   = (float*)(smem + O_DWB);
    _Float16* outh  = (_Float16*)(smem + O_OUT);
    float*    kmean = (float*)(smem + O_KM);   // also rsc during P0..P1
    float*    rsc   = (float*)(smem + O_KM);
    float*    zbuf  = (float*)(smem + O_Z);

    int tid  = threadIdx.x;
    int lane = tid & 31;
    int wave = tid >> 5;
    int w    = blockIdx.x;          // 4096 windows
    int bimg = w >> 8;
    int wr   = (w >> 4) & 15;
    int wc   = w & 15;

    // Warm L2 with the f16 weights used by this kernel (global_prefetch_b8).
    for (int i = tid * 128; i < 288 * 96 * 2; i += 256 * 128)
        __builtin_prefetch((const char*)Wqkv + i, 0, 1);
    for (int i = tid * 128; i < 96 * 96 * 2; i += 256 * 128)
        __builtin_prefetch((const char*)Wproj + i, 0, 1);

    // P0: stage rolled x rows (window tokens) into qf (f32), rows >= 49 zero.
    // Also precompute rsc[c] = 1/softplus(scale_p[c]) once (hoisted out of P1).
    if (tid < 96) {
        float sp = scale_p[tid];
        float sc = (sp > 20.f) ? sp : log1pf(expf(sp));
        rsc[tid] = 1.f / sc;
    }
    for (int i = tid; i < 64 * 96; i += 256) {
        int t = i / 96, c = i % 96;
        float v = 0.f;
        if (t < NTOK) {
            int rr = t / WSZ, cl = t % WSZ;
            int ho = (wr * WSZ + rr + SHFT) % HH;
            int wo = (wc * WSZ + cl + SHFT) % WWI;
            v = x[((size_t)bimg * (HH * WWI) + (size_t)ho * WWI + wo) * CC + c];
        }
        qf[i] = v;
    }
    __syncthreads();

    // P0b: LayerNorm1 -> xwin (f16), rows >= 49 zero
    if (tid < 64) {
        int t = tid;
        float mu = 0.f, s2 = 0.f;
        for (int c = 0; c < 96; c++) { float v = qf[t * 96 + c]; mu += v; s2 += v * v; }
        mu /= 96.f;
        float inv = rsqrtf(s2 / 96.f - mu * mu + 1e-5f);
        for (int c = 0; c < 96; c++) {
            float v = (t < NTOK) ? ((qf[t * 96 + c] - mu) * inv * n1g[c] + n1b[c]) : 0.f;
            xwin[t * 96 + c] = (_Float16)v;
        }
    }
    __syncthreads();

    // P1: qkv GEMM (64x96 @ 96x288). q,k -> LDS f32 with relu/eps/scale; v in regs.
    v8f vacc[3];
    for (int j = 0; j < 6; j++) {
        int s = wave + j * 8;               // 48 q/k subtiles
        int mi = s & 3, ni = s >> 2;        // ni 0..11 (cols 0..191)
        v8f acc = {};
        for (int kk = 0; kk < 96; kk += 32) {
            v16h a = ld_a_rm(xwin, 96, mi * 16, kk, lane);
            v16h b = ld_b_wt(Wqkv, 96, ni * 16, kk, lane);
            acc = wmma16(a, b, acc);
        }
        int col = ni * 16 + (lane & 15);
        float bias = bqkv[col];
        int isq = col < 96;
        int cc = isq ? col : col - 96;
        float rs = rsc[cc];
        float* dst = isq ? qf : kf;
        int g = lane >> 4;
#pragma unroll
        for (int r = 0; r < 8; r++) {
            int m = mi * 16 + r + 8 * g;
            dst[m * 96 + cc] = (fmaxf(acc[r] + bias, 0.f) + FEPS) * rs;
        }
    }
    for (int j = 0; j < 3; j++) {
        int s = wave + j * 8;               // 24 v subtiles
        int mi = s & 3, ni = s >> 2;        // ni 0..5
        v8f acc = {};
        for (int kk = 0; kk < 96; kk += 32) {
            v16h a = ld_a_rm(xwin, 96, mi * 16, kk, lane);
            v16h b = ld_b_wt(Wqkv, 96, 192 + ni * 16, kk, lane);
            acc = wmma16(a, b, acc);
        }
        vacc[j] = acc;
    }
    __syncthreads();

    // P2: focused q transform: q = q^3 / ||q^3|| * ||q||  -> qh (f16) over dead xwin
    if (tid < 64) {
        int t = tid;
        float s2 = 0.f, s6 = 0.f;
        for (int c = 0; c < 96; c++) {
            float v = qf[t * 96 + c]; s2 += v * v;
            float v3 = v * v * v; s6 += v3 * v3;
        }
        float inv = sqrtf(s2) * rsqrtf(s6);
        for (int c = 0; c < 96; c++) {
            float v = qf[t * 96 + c];
            float v3 = v * v * v;
            qh[t * 96 + c] = (t < NTOK) ? (_Float16)(v3 * inv) : (_Float16)0.f;
        }
    }
    __syncthreads();

    // P3: k transform -> kh (over dead qf front), v regs -> vh (over dead qf back)
    if (tid < 64) {
        int t = tid;
        float s2 = 0.f, s6 = 0.f;
        for (int c = 0; c < 96; c++) {
            float v = kf[t * 96 + c]; s2 += v * v;
            float v3 = v * v * v; s6 += v3 * v3;
        }
        float inv = sqrtf(s2) * rsqrtf(s6);
        for (int c = 0; c < 96; c++) {
            float v = kf[t * 96 + c];
            float v3 = v * v * v;
            kh[t * 96 + c] = (t < NTOK) ? (_Float16)(v3 * inv) : (_Float16)0.f;
        }
    }
    {
        int g = lane >> 4;
        for (int j = 0; j < 3; j++) {
            int s = wave + j * 8;
            int mi = s & 3, ni = s >> 2;
            int cc = ni * 16 + (lane & 15);
            float bias = bqkv[192 + cc];
#pragma unroll
            for (int r = 0; r < 8; r++) {
                int m = mi * 16 + r + 8 * g;
                float v = (m < NTOK) ? (vacc[j][r] + bias) : 0.f;
                vh[m * 96 + cc] = (_Float16)v;
            }
        }
    }
    __syncthreads();

    // P4: kmean, dwc weight load, kv = (k^T v)/N via WMMA (over dead kf region)
    if (tid < 96) {
        float s = 0.f;
        for (int n = 0; n < NTOK; n++) s += (float)kh[n * 96 + tid];
        kmean[tid] = s / (float)NTOK;
    }
    for (int i = tid; i < HDIM * 25; i += 256) wdw[i] = dwc_w[i];
    if (tid < HDIM) bdw[tid] = dwc_b[tid];
    for (int s = wave; s < 12; s += 8) {       // 3 heads x 2x2 subtiles
        int head = s >> 2, mi = (s >> 1) & 1, ni = s & 1;
        v8f acc = {};
        for (int kk = 0; kk < 64; kk += 32) {  // contract over tokens (rows>=49 are zero)
            v16h a = ld_a_tr(kh, 96, head * 32 + mi * 16, kk, lane);
            v16h b = ld_b_rm(vh, 96, head * 32 + ni * 16, kk, lane);
            acc = wmma16(a, b, acc);
        }
        int g = lane >> 4, n = ni * 16 + (lane & 15);
#pragma unroll
        for (int r = 0; r < 8; r++) {
            int m = mi * 16 + r + 8 * g;
            kvh[head * 1024 + m * 32 + n] = (_Float16)(acc[r] * (1.f / (float)NTOK));
        }
    }
    __syncthreads();

    // P5: z = 1/(q . kmean + eps)
    if (tid < 192) {
        int t = tid & 63, head = tid >> 6;
        float d = 0.f;
        for (int c = 0; c < 32; c++)
            d += (float)qh[t * 96 + head * 32 + c] * kmean[head * 32 + c];
        zbuf[head * 64 + t] = 1.f / (d + FEPS);
    }
    __syncthreads();

    // P6: out = (q @ kv) * z + depthwise-conv(v)  -> outh (f16)
    for (int s = wave; s < 24; s += 8) {
        int mi = s & 3, nt = s >> 2;           // nt 0..5
        int head = nt >> 1, nn = (nt & 1) * 16;
        v16h a = ld_a_rm(qh, 96, mi * 16, head * 32, lane);
        v16h b = ld_b_rm(kvh + head * 1024, 32, nn, 0, lane);
        v8f acc = {};
        acc = wmma16(a, b, acc);
        int g = lane >> 4;
        int c = head * 32 + nn + (lane & 15);
        int d = c & 31;
#pragma unroll
        for (int r = 0; r < 8; r++) {
            int m = mi * 16 + r + 8 * g;
            float v = 0.f;
            if (m < NTOK) {
                float dv = bdw[d];
                int rr = m / WSZ, cl = m % WSZ;
                for (int ky = 0; ky < 5; ky++) {
                    int yy = rr + ky - 2;
                    if (yy < 0 || yy >= WSZ) continue;
                    for (int kx = 0; kx < 5; kx++) {
                        int xx = cl + kx - 2;
                        if (xx < 0 || xx >= WSZ) continue;
                        dv += (float)vh[(yy * WSZ + xx) * 96 + c] * wdw[d * 25 + ky * 5 + kx];
                    }
                }
                v = acc[r] * zbuf[head * 64 + m] + dv;
            }
            outh[m * 96 + c] = (_Float16)v;
        }
    }
    __syncthreads();

    // P7: proj GEMM + window-reverse + roll-back + residual -> x2
    for (int s = wave; s < 24; s += 8) {
        int mi = s & 3, nt = s >> 2;
        v8f acc = {};
        for (int kk = 0; kk < 96; kk += 32) {
            v16h a = ld_a_rm(outh, 96, mi * 16, kk, lane);
            v16h b = ld_b_wt(Wproj, 96, nt * 16, kk, lane);
            acc = wmma16(a, b, acc);
        }
        int g = lane >> 4, n = nt * 16 + (lane & 15);
        float bias = bproj[n];
#pragma unroll
        for (int r = 0; r < 8; r++) {
            int m = mi * 16 + r + 8 * g;
            if (m < NTOK) {
                int rr = m / WSZ, cl = m % WSZ;
                int ho = (wr * WSZ + rr + SHFT) % HH;
                int wo = (wc * WSZ + cl + SHFT) % WWI;
                size_t idx = ((size_t)bimg * (HH * WWI) + (size_t)ho * WWI + wo) * CC + n;
                x2[idx] = x[idx] + acc[r] + bias;
            }
        }
    }
}

// ---------------- Fused LN2 + MLP kernel (61440 B LDS) -----------------------
#define M_Y   0        // y f16 64x96 (12288)
#define M_H   12288    // hidden f16 64x384 (49152)
#define M_ST  12288    // staging f32 64x96 (24576, dead before fc1 writes hidden)
#define SMEM_MLP 61440

__global__ __launch_bounds__(256)
void mlp_kernel(const float* __restrict__ x2,
                const float* __restrict__ n2g, const float* __restrict__ n2b,
                const _Float16* __restrict__ W1, const float* __restrict__ b1,
                const _Float16* __restrict__ W2, const float* __restrict__ b2,
                float* __restrict__ out) {
    __shared__ char smem[SMEM_MLP];
    _Float16* yh  = (_Float16*)(smem + M_Y);
    _Float16* hid = (_Float16*)(smem + M_H);
    float*    stg = (float*)(smem + M_ST);

    int tid  = threadIdx.x;
    int lane = tid & 31;
    int wave = tid >> 5;
    size_t t0 = (size_t)blockIdx.x * 64;

    // Warm L2 with the f16 MLP weights (global_prefetch_b8).
    for (int i = tid * 128; i < HIDN * 96 * 2; i += 256 * 128) {
        __builtin_prefetch((const char*)W1 + i, 0, 1);
        __builtin_prefetch((const char*)W2 + i, 0, 1);
    }

    for (int i = tid; i < 64 * 96; i += 256) stg[i] = x2[t0 * 96 + i];
    __syncthreads();

    if (tid < 64) {
        int t = tid;
        float mu = 0.f, s2 = 0.f;
        for (int c = 0; c < 96; c++) { float v = stg[t * 96 + c]; mu += v; s2 += v * v; }
        mu /= 96.f;
        float inv = rsqrtf(s2 / 96.f - mu * mu + 1e-5f);
        for (int c = 0; c < 96; c++)
            yh[t * 96 + c] = (_Float16)((stg[t * 96 + c] - mu) * inv * n2g[c] + n2b[c]);
    }
    __syncthreads();

    // fc1 (64x96 @ 96x384) + exact GELU
    for (int s = wave; s < 96; s += 8) {
        int mi = s & 3, nt = s >> 2;           // nt 0..23
        v8f acc = {};
        for (int kk = 0; kk < 96; kk += 32) {
            v16h a = ld_a_rm(yh, 96, mi * 16, kk, lane);
            v16h b = ld_b_wt(W1, 96, nt * 16, kk, lane);
            acc = wmma16(a, b, acc);
        }
        int g = lane >> 4, n = nt * 16 + (lane & 15);
        float bias = b1[n];
#pragma unroll
        for (int r = 0; r < 8; r++) {
            int m = mi * 16 + r + 8 * g;
            float v = acc[r] + bias;
            v = 0.5f * v * (1.f + erff(v * 0.70710678118f));
            hid[m * HIDN + n] = (_Float16)v;
        }
    }
    __syncthreads();

    // fc2 (64x384 @ 384x96) + residual
    for (int s = wave; s < 24; s += 8) {
        int mi = s & 3, nt = s >> 2;
        v8f acc = {};
        for (int kk = 0; kk < HIDN; kk += 32) {
            v16h a = ld_a_rm(hid, HIDN, mi * 16, kk, lane);
            v16h b = ld_b_wt(W2, HIDN, nt * 16, kk, lane);
            acc = wmma16(a, b, acc);
        }
        int g = lane >> 4, n = nt * 16 + (lane & 15);
        float bias = b2[n];
#pragma unroll
        for (int r = 0; r < 8; r++) {
            int m = mi * 16 + r + 8 * g;
            size_t idx = (t0 + m) * 96 + n;
            out[idx] = x2[idx] + acc[r] + bias;
        }
    }
}

// ---------------- f32 -> f16 weight conversion -------------------------------
__global__ void cvt_kernel(const float* __restrict__ a, _Float16* __restrict__ o, int n) {
    int i = blockIdx.x * 256 + threadIdx.x;
    if (i < n) o[i] = (_Float16)a[i];
}

extern "C" void kernel_launch(void* const* d_in, const int* in_sizes, int n_in,
                              void* d_out, int out_size, void* d_ws, size_t ws_size,
                              hipStream_t stream) {
    const float* x       = (const float*)d_in[0];
    const float* n1g     = (const float*)d_in[1];
    const float* n1b     = (const float*)d_in[2];
    const float* qkv_w   = (const float*)d_in[3];
    const float* qkv_b   = (const float*)d_in[4];
    const float* scale_p = (const float*)d_in[5];
    const float* dwc_w   = (const float*)d_in[6];
    const float* dwc_b   = (const float*)d_in[7];
    const float* proj_w  = (const float*)d_in[8];
    const float* proj_b  = (const float*)d_in[9];
    const float* n2g     = (const float*)d_in[10];
    const float* n2b     = (const float*)d_in[11];
    const float* fc1_w   = (const float*)d_in[12];
    const float* fc1_b   = (const float*)d_in[13];
    const float* fc2_w   = (const float*)d_in[14];
    const float* fc2_b   = (const float*)d_in[15];

    char* ws = (char*)d_ws;
    float* x2 = (float*)ws;                              // 16*112*112*96 f32
    size_t off = (size_t)BB * HH * WWI * CC * sizeof(float);   // 77070336 (32B aligned)
    _Float16* Wqkv  = (_Float16*)(ws + off); off += (size_t)288 * 96 * 2;
    _Float16* Wproj = (_Float16*)(ws + off); off += (size_t)96 * 96 * 2;
    _Float16* W1    = (_Float16*)(ws + off); off += (size_t)384 * 96 * 2;
    _Float16* W2    = (_Float16*)(ws + off); off += (size_t)96 * 384 * 2;

    cvt_kernel<<<(288 * 96 + 255) / 256, 256, 0, stream>>>(qkv_w, Wqkv, 288 * 96);
    cvt_kernel<<<(96 * 96 + 255) / 256, 256, 0, stream>>>(proj_w, Wproj, 96 * 96);
    cvt_kernel<<<(384 * 96 + 255) / 256, 256, 0, stream>>>(fc1_w, W1, 384 * 96);
    cvt_kernel<<<(96 * 384 + 255) / 256, 256, 0, stream>>>(fc2_w, W2, 96 * 384);

    // one workgroup per 7x7 window: 16 imgs * 16*16 windows = 4096
    attn_kernel<<<4096, 256, 0, stream>>>(x, n1g, n1b, Wqkv, qkv_b, scale_p,
                                          dwc_w, dwc_b, Wproj, proj_b, x2);
    // one workgroup per 64 tokens: 200704/64 = 3136
    mlp_kernel<<<(BB * HH * WWI) / 64, 256, 0, stream>>>(x2, n2g, n2b, W1, fc1_b,
                                                         W2, fc2_b, (float*)d_out);
}
